// Pnet_34548716929429
// MI455X (gfx1250) — matrix operand order repaired and checked by reading, not compiled, not run on hardware
//
#include <hip/hip_runtime.h>
#include <math.h>

typedef __attribute__((ext_vector_type(16))) _Float16 v16h;
typedef __attribute__((ext_vector_type(8)))  _Float16 v8h;
typedef __attribute__((ext_vector_type(8)))  float    v8f;
typedef __attribute__((ext_vector_type(4)))  unsigned int v4u;
typedef __attribute__((ext_vector_type(4)))  int v4i_t;
typedef __attribute__((ext_vector_type(8)))  int v8i_t;

#define TKN  4096
#define GTOT 1536
#define KEYMAX (~0ull)

// ---------------------------------------------------------------------------
// A-operand gather (16-bit A 16x32), contiguous k order matching B_swz:
// lane l: m = l&15, kh = l>>4 ; element e: K = e + 16*kh
// ---------------------------------------------------------------------------
static __device__ inline v16h gather_A(const _Float16* row, int kh, int kbase) {
  v8h lo = *(const v8h*)(row + kbase + 16 * kh);
  v8h hi = *(const v8h*)(row + kbase + 16 * kh + 8);
  v16h a;
#pragma unroll
  for (int i = 0; i < 8; ++i) { a[i] = lo[i]; a[i + 8] = hi[i]; }
  return a;
}

// ---------------------------------------------------------------------------
// TDM: 1-D tile load global->LDS (ndw dwords), D# packed per ISA 8.3/8.4.
// group0: [1:0]=count=1 | lds_addr | global_addr[56:0] | type=2
// group1: data_size=4B, tensor_dim0=tile_dim0=ndw, tensor/tile_dim1=1
// ---------------------------------------------------------------------------
static __device__ inline void tdm_load_1d(unsigned lds_off, unsigned long long gaddr,
                                          unsigned ndw) {
  v4u g0;
  g0[0] = 1u;                                                    // count=1
  g0[1] = lds_off;                                               // lds_addr
  g0[2] = (unsigned)(gaddr & 0xFFFFFFFFull);                     // ga[31:0]
  g0[3] = (unsigned)((gaddr >> 32) & 0x1FFFFFFull) | (2u << 30); // ga[56:32]|type=2
  v8i_t g1;
  g1[0] = (int)(2u << 16);                       // wg_mask=0, data_size=4B
  g1[1] = (int)((ndw & 0xFFFFu) << 16);          // tensor_dim0[15:0]
  g1[2] = (int)((ndw >> 16) | (1u << 16));       // tensor_dim0[31:16] | tensor_dim1=1
  g1[3] = (int)((ndw & 0xFFFFu) << 16);          // tile_dim0 = ndw
  g1[4] = 1;                                     // tile_dim1 = 1
  g1[5] = (int)ndw;                              // tensor_dim0_stride
  g1[6] = 0;
  g1[7] = 0;
  v4i_t z4 = {0, 0, 0, 0};
#if defined(__clang_major__) && (__clang_major__ >= 23)
  v8i_t z8 = {0, 0, 0, 0, 0, 0, 0, 0};
  __builtin_amdgcn_tensor_load_to_lds(g0, g1, z4, z4, z8, 0);
#else
  __builtin_amdgcn_tensor_load_to_lds(g0, g1, z4, z4, 0);
#endif
}

static __device__ inline void bitonic4096(unsigned long long* sh, int tid, int nth) {
  for (int size = 2; size <= TKN; size <<= 1)
    for (int stride = size >> 1; stride > 0; stride >>= 1) {
      __syncthreads();
      for (int i = tid; i < TKN; i += nth) {
        int p = i ^ stride;
        if (p > i) {
          unsigned long long a = sh[i], b = sh[p];
          bool asc = ((i & size) == 0);
          if ((a > b) == asc) { sh[i] = b; sh[p] = a; }
        }
      }
    }
  __syncthreads();
}

// ---------------------------------------------------------------------------
__global__ void k_zero(float* p, int n) {
  int i = blockIdx.x * blockDim.x + threadIdx.x;
  if (i < n) p[i] = 0.0f;
}

// Pre-swizzle B matrices into per-lane WMMA layout:
// B_swz[step][lane][e] = B[step*32 + (lane>>4)*16 + e][lane&15]
__global__ void k_prep(const float* __restrict__ w2, const float* __restrict__ w3,
                       const float* __restrict__ wa, const float* __restrict__ wb,
                       _Float16* b2sw, _Float16* b3sw, _Float16* bhsw) {
  int stride = gridDim.x * blockDim.x;
  int tid = blockIdx.x * blockDim.x + threadIdx.x;
  for (int t = tid; t < 3 * 32 * 16; t += stride) {          // conv2: K=90, N=16
    int s = t / 512, l = (t % 512) / 16, e = t % 16;
    int k = s * 32 + (l >> 4) * 16 + e, n = l & 15;
    float v = 0.f;
    if (k < 90) {
      int c = k / 9, r9 = k % 9, ky = r9 / 3, kx = r9 % 3;
      v = w2[((n * 10 + c) * 3 + ky) * 3 + kx];
    }
    b2sw[t] = (_Float16)v;
  }
  for (int t = tid; t < 2 * 5 * 32 * 16; t += stride) {      // conv3: K=144, N=32
    int nh = t / 2560, rem = t % 2560;
    int s = rem / 512, l = (rem % 512) / 16, e = t % 16;
    int k = s * 32 + (l >> 4) * 16 + e, n = l & 15, ch = nh * 16 + n;
    float v = 0.f;
    if (k < 144) {
      int c = k / 9, r9 = k % 9, ky = r9 / 3, kx = r9 % 3;
      v = w3[((ch * 16 + c) * 3 + ky) * 3 + kx];
    }
    b3sw[t] = (_Float16)v;
  }
  for (int t = tid; t < 32 * 16; t += stride) {              // heads: K=32, N=6
    int l = t / 16, e = t % 16;
    int k = (l >> 4) * 16 + e, n = l & 15;
    float v = 0.f;
    if (n == 0) v = wa[k];
    else if (n == 1) v = wa[32 + k];
    else if (n < 6) v = wb[(n - 2) * 32 + k];
    bhsw[t] = (_Float16)v;
  }
}

// conv1(3->10) + PReLU + maxpool2, nearest-resize fused into the input fetch.
__global__ void k_conv1pool(const float* __restrict__ img, const float* __restrict__ w1,
                            const float* __restrict__ b1, const float* __restrict__ a1,
                            _Float16* __restrict__ P, int ph, int pw, float rh, float rw) {
  int t = blockIdx.x * blockDim.x + threadIdx.x;
  if (t >= ph * pw) return;
  int py = t / pw, px = t % pw;
  float in[4][4][3];
#pragma unroll
  for (int dy = 0; dy < 4; dy++) {
    int iy = 2 * py + dy;
    int ri = (int)((float)iy * rh); if (ri > 1079) ri = 1079;
#pragma unroll
    for (int dx = 0; dx < 4; dx++) {
      int ix = 2 * px + dx;
      int ci = (int)((float)ix * rw); if (ci > 1919) ci = 1919;
      const float* p = img + ((size_t)ri * 1920 + ci) * 3;
#pragma unroll
      for (int ch = 0; ch < 3; ch++) in[dy][dx][ch] = (p[ch] - 127.5f) * 0.0078125f;
    }
  }
  for (int oc = 0; oc < 10; ++oc) {
    const float* w = w1 + oc * 27;
    float bb = b1[oc], aa = a1[oc], mx = -1e30f;
#pragma unroll
    for (int wy = 0; wy < 2; wy++)
#pragma unroll
      for (int wx = 0; wx < 2; wx++) {
        float acc = bb;
#pragma unroll
        for (int c = 0; c < 3; c++)
#pragma unroll
          for (int ky = 0; ky < 3; ky++)
#pragma unroll
            for (int kx = 0; kx < 3; kx++)
              acc += w[(c * 3 + ky) * 3 + kx] * in[wy + ky][wx + kx][c];
        float v = acc > 0.f ? acc : aa * acc;
        mx = fmaxf(mx, v);
      }
    P[((size_t)oc * ph + py) * pw + px] = (_Float16)mx;
  }
}

// conv2 (10->16) WMMA implicit GEMM. 256 threads = 8 waves = 8 tiles of 16 px.
// B (3KB) staged once per block into LDS by TDM, waited on TENSORcnt.
__global__ void k_conv2(const _Float16* __restrict__ P, const _Float16* __restrict__ Bsw,
                        const float* __restrict__ b2, const float* __restrict__ a2,
                        _Float16* __restrict__ C2, int ph, int pw, int c2h, int c2w,
                        int ntx) {
  __shared__ __align__(16) _Float16 AtAll[8 * 16 * 96];
  __shared__ __align__(16) _Float16 Bl[3 * 512];
  int tid = threadIdx.x, wv = tid >> 5, lane = tid & 31;
  _Float16* At = AtAll + wv * 16 * 96;
  int tile = blockIdx.x * 8 + wv;
  int tx = tile % ntx, ty = tile / ntx;
  int x0 = tx * 16, y = ty;
  bool live = (ty < c2h);
  for (int t = lane; t < 16 * 96; t += 32) {
    int m = t / 96, k = t % 96;
    float v = 0.f;
    if (live && k < 90) {
      int c = k / 9, r9 = k % 9, dy = r9 / 3, dx = r9 % 3;
      int xx = x0 + m + dx;
      if (xx < pw) v = (float)P[((size_t)c * ph + (y + dy)) * pw + xx];
    }
    At[m * 96 + k] = (_Float16)v;
  }
  if (wv == 0) {
    tdm_load_1d((unsigned)(unsigned long long)(void*)Bl,
                (unsigned long long)(void*)Bsw, (3 * 512 * 2) / 4);
    __builtin_amdgcn_s_wait_tensorcnt(0);
  }
  __syncthreads();
  int mlo = lane & 15, kh = lane >> 4;
  v8f acc = {0.f, 0.f, 0.f, 0.f, 0.f, 0.f, 0.f, 0.f};
#pragma unroll
  for (int s = 0; s < 3; ++s) {
    v16h a = gather_A(At + mlo * 96, kh, s * 32);
    v16h b = *(const v16h*)(Bl + (s * 32 + lane) * 16);
    acc = __builtin_amdgcn_wmma_f32_16x16x32_f16(false, a, false, b, (short)0, acc, false, false);
  }
  int n = lane & 15;
  float bb = b2[n], aa = a2[n];
  if (live && x0 + 16 <= c2w) {
#pragma unroll
    for (int r = 0; r < 8; ++r) {
      int m = r + 8 * kh;
      float v = acc[r] + bb;
      v = v > 0.f ? v : aa * v;
      C2[((size_t)n * c2h + y) * c2w + x0 + m] = (_Float16)v;
    }
  } else if (live) {
#pragma unroll
    for (int r = 0; r < 8; ++r) {
      int m = r + 8 * kh, x = x0 + m;
      if (x < c2w) {
        float v = acc[r] + bb;
        v = v > 0.f ? v : aa * v;
        C2[((size_t)n * c2h + y) * c2w + x] = (_Float16)v;
      }
    }
  }
}

// conv3 (16->32, K=144->160) + 1x1 heads (K=32) + softmax, fused.
// 8 waves/block; B3+Bh blob (11KB) staged once per block via TDM.
__global__ void k_conv3heads(const _Float16* __restrict__ C2, const _Float16* __restrict__ B3sw,
                             const float* __restrict__ b3, const float* __restrict__ a3,
                             const float* __restrict__ ba, const float* __restrict__ bb,
                             float* __restrict__ prob, float* __restrict__ reg,
                             int c2h, int c2w, int c3h, int c3w, int ntx) {
  __shared__ __align__(16) _Float16 AtAll[8 * 16 * 160];
  __shared__ __align__(16) _Float16 TrAll[8 * 16 * 32];
  __shared__ float HoAll[8 * 16 * 8];
  __shared__ __align__(16) _Float16 Bl[5632];   // [0,5120)=conv3 B, [5120,5632)=head B
  int tid = threadIdx.x, wv = tid >> 5, lane = tid & 31;
  _Float16* At = AtAll + wv * 16 * 160;
  _Float16* Tr = TrAll + wv * 16 * 32;
  float* Ho = HoAll + wv * 16 * 8;
  int tile = blockIdx.x * 8 + wv;
  int tx = tile % ntx, ty = tile / ntx;
  int x0 = tx * 16, y = ty;
  bool live = (ty < c3h);
  for (int t = lane; t < 16 * 160; t += 32) {
    int m = t / 160, k = t % 160;
    float v = 0.f;
    if (live && k < 144) {
      int c = k / 9, r9 = k % 9, dy = r9 / 3, dx = r9 % 3;
      int xx = x0 + m + dx;
      if (xx < c2w) v = (float)C2[((size_t)c * c2h + (y + dy)) * c2w + xx];
    }
    At[m * 160 + k] = (_Float16)v;
  }
  if (wv == 0) {
    tdm_load_1d((unsigned)(unsigned long long)(void*)Bl,
                (unsigned long long)(void*)B3sw, (5632 * 2) / 4);
    __builtin_amdgcn_s_wait_tensorcnt(0);
  }
  __syncthreads();
  int mlo = lane & 15, kh = lane >> 4, n = lane & 15;
  v8f a0 = {0.f, 0.f, 0.f, 0.f, 0.f, 0.f, 0.f, 0.f};
  v8f a1 = {0.f, 0.f, 0.f, 0.f, 0.f, 0.f, 0.f, 0.f};
#pragma unroll
  for (int s = 0; s < 5; ++s) {
    v16h a = gather_A(At + mlo * 160, kh, s * 32);
    v16h blv = *(const v16h*)(Bl + ((0 * 5 + s) * 32 + lane) * 16);
    v16h bhv = *(const v16h*)(Bl + ((1 * 5 + s) * 32 + lane) * 16);
    a0 = __builtin_amdgcn_wmma_f32_16x16x32_f16(false, a, false, blv, (short)0, a0, false, false);
    a1 = __builtin_amdgcn_wmma_f32_16x16x32_f16(false, a, false, bhv, (short)0, a1, false, false);
  }
  float b3l = b3[n], a3l = a3[n], b3h = b3[16 + n], a3h = a3[16 + n];
#pragma unroll
  for (int r = 0; r < 8; ++r) {
    int m = r + 8 * kh;
    float v0 = a0[r] + b3l; v0 = v0 > 0.f ? v0 : a3l * v0;
    float v1 = a1[r] + b3h; v1 = v1 > 0.f ? v1 : a3h * v1;
    Tr[m * 32 + n]      = (_Float16)v0;   // pixel-major [m][channel]
    Tr[m * 32 + 16 + n] = (_Float16)v1;
  }
  __syncthreads();
  v16h ah = gather_A(Tr + mlo * 32, kh, 0);
  v16h bh = *(const v16h*)(Bl + 5120 + lane * 16);
  v8f hacc = {0.f, 0.f, 0.f, 0.f, 0.f, 0.f, 0.f, 0.f};
  hacc = __builtin_amdgcn_wmma_f32_16x16x32_f16(false, ah, false, bh, (short)0, hacc, false, false);
  float hb = 0.f;
  if (n == 0) hb = ba[0]; else if (n == 1) hb = ba[1]; else if (n < 6) hb = bb[n - 2];
#pragma unroll
  for (int r = 0; r < 8; ++r) {
    int m = r + 8 * kh;
    if (n < 6) Ho[m * 8 + n] = hacc[r] + hb;
  }
  __syncthreads();
  if (lane < 16 && live) {
    int x = x0 + lane;
    if (x < c3w) {
      int pix = y * c3w + x;
      float c0 = Ho[lane * 8 + 0], c1 = Ho[lane * 8 + 1];
      float mx = fmaxf(c0, c1);
      float e0 = expf(c0 - mx), e1 = expf(c1 - mx);
      prob[pix] = e1 / (e0 + e1);
      reg[(size_t)pix * 4 + 0] = Ho[lane * 8 + 2];
      reg[(size_t)pix * 4 + 1] = Ho[lane * 8 + 3];
      reg[(size_t)pix * 4 + 2] = Ho[lane * 8 + 4];
      reg[(size_t)pix * 4 + 3] = Ho[lane * 8 + 5];
    }
  }
}

// One top-k reduction stage: each block sorts a 4096 chunk, keeps its top-512.
// key = (~score_bits)<<32 | idx  -> ascending sort == (score desc, idx asc).
__global__ void k_topk_stage(const float* prob, const unsigned long long* kin,
                             unsigned long long* kout, int m) {
  __shared__ unsigned long long sh[TKN];
  int tid = threadIdx.x, base = blockIdx.x * TKN;
  for (int i = tid; i < TKN; i += blockDim.x) {
    int gi = base + i;
    unsigned long long key = KEYMAX;
    if (gi < m) {
      if (prob) {
        float p = prob[gi];
        if (p >= 0.55f)
          key = ((unsigned long long)(~__float_as_uint(p)) << 32) | (unsigned)gi;
      } else key = kin[gi];
    }
    sh[i] = key;
  }
  bitonic4096(sh, tid, blockDim.x);
  for (int i = tid; i < 512; i += blockDim.x) kout[blockIdx.x * 512 + i] = sh[i];
}

// Per-scale greedy NMS (IOU 0.5) on the sorted top-K, stable-partition kept
// entries into the first KK per-scale result slots.
__global__ void k_nms_scale(const unsigned long long* __restrict__ topk,
                            const float* __restrict__ reg, float s, int gw, int K, int KK,
                            int scbase, int* resFlag, float* resScore,
                            float* resBox, float* resReg) {
  __shared__ float bx[512][4];
  __shared__ int keep[512];
  __shared__ int slots[512];
  __shared__ unsigned long long skey[512];
  int tid = threadIdx.x;
  unsigned long long key = (tid < K) ? topk[tid] : KEYMAX;
  skey[tid] = key;
  int valid = (key != KEYMAX);
  float b0 = 0.f, b1 = 0.f, b2 = 0.f, b3 = 0.f;
  if (valid) {
    unsigned idx = (unsigned)key;
    float yy = (float)(idx / (unsigned)gw), xx = (float)(idx % (unsigned)gw);
    b0 = (yy * 2.f + 1.f) / s;  b1 = (xx * 2.f + 1.f) / s;
    b2 = (yy * 2.f + 12.f) / s; b3 = (xx * 2.f + 12.f) / s;
  }
  bx[tid][0] = b0; bx[tid][1] = b1; bx[tid][2] = b2; bx[tid][3] = b3;
  keep[tid] = valid;
  for (int i = 0; i < K; i++) {
    __syncthreads();
    if (!keep[i]) continue;
    float ax1 = bx[i][0], ay1 = bx[i][1], ax2 = bx[i][2], ay2 = bx[i][3];
    if (tid > i && keep[tid]) {
      float areaA = (ax2 - ax1) * (ay2 - ay1);
      float areaB = (bx[tid][2] - bx[tid][0]) * (bx[tid][3] - bx[tid][1]);
      float iw = fmaxf(fminf(ax2, bx[tid][2]) - fmaxf(ax1, bx[tid][0]), 0.f);
      float ih = fmaxf(fminf(ay2, bx[tid][3]) - fmaxf(ay1, bx[tid][1]), 0.f);
      float inter = iw * ih;
      float iou = inter / (areaA + areaB - inter + 1e-9f);
      if (iou > 0.5f) keep[tid] = 0;
    }
  }
  __syncthreads();
  if (tid == 0) {
    int nk = 0;
    for (int j = 0; j < K; j++) slots[j] = keep[j] ? nk++ : 0x7fffffff;
  }
  __syncthreads();
  if (tid < K && keep[tid] && slots[tid] < KK) {
    int g = scbase + slots[tid];
    unsigned long long k2 = skey[tid];
    unsigned idx = (unsigned)k2;
    resFlag[g] = 1;
    resScore[g] = __uint_as_float(~(unsigned)(k2 >> 32));
    resBox[g * 4 + 0] = bx[tid][0]; resBox[g * 4 + 1] = bx[tid][1];
    resBox[g * 4 + 2] = bx[tid][2]; resBox[g * 4 + 3] = bx[tid][3];
    const float* rg = reg + (size_t)idx * 4;
    resReg[g * 4 + 0] = rg[0]; resReg[g * 4 + 1] = rg[1];
    resReg[g * 4 + 2] = rg[2]; resReg[g * 4 + 3] = rg[3];
  }
}

__global__ void k_gsort(const int* __restrict__ resFlag, const float* __restrict__ resScore,
                        unsigned long long* gkeys) {
  __shared__ unsigned long long sh[TKN];
  int tid = threadIdx.x;
  for (int i = tid; i < TKN; i += blockDim.x) {
    unsigned long long key = KEYMAX;
    if (i < GTOT && resFlag[i])
      key = ((unsigned long long)(~__float_as_uint(resScore[i])) << 32) | (unsigned)i;
    sh[i] = key;
  }
  bitonic4096(sh, tid, blockDim.x);
  for (int i = tid; i < GTOT; i += blockDim.x) gkeys[i] = sh[i];
}

// Global NMS (IOU 0.7) over all scales + box refinement + squaring + output.
__global__ void k_gnms(const unsigned long long* __restrict__ gkeys,
                       const float* __restrict__ resBox, const float* __restrict__ resReg,
                       float* __restrict__ out, int NOUT) {
  __shared__ float sb[GTOT * 4];
  __shared__ int skeep[GTOT];
  __shared__ int sgid[GTOT];
  __shared__ float ssc[GTOT];
  int tid = threadIdx.x;
  for (int j = tid; j < GTOT; j += blockDim.x) {
    unsigned long long key = gkeys[j];
    int v = (key != KEYMAX);
    int g = v ? (int)(unsigned)key : 0;
    sgid[j] = g;
    ssc[j] = v ? __uint_as_float(~(unsigned)(key >> 32)) : 0.f;
    skeep[j] = v;
    for (int q = 0; q < 4; q++) sb[j * 4 + q] = v ? resBox[g * 4 + q] : 0.f;
  }
  for (int i = 0; i < GTOT; i++) {
    __syncthreads();
    if (!skeep[i]) continue;
    float ax1 = sb[i * 4], ay1 = sb[i * 4 + 1], ax2 = sb[i * 4 + 2], ay2 = sb[i * 4 + 3];
    float areaA = (ax2 - ax1) * (ay2 - ay1);
    for (int j = tid; j < GTOT; j += blockDim.x) {
      if (j > i && skeep[j]) {
        float areaB = (sb[j * 4 + 2] - sb[j * 4]) * (sb[j * 4 + 3] - sb[j * 4 + 1]);
        float iw = fmaxf(fminf(ax2, sb[j * 4 + 2]) - fmaxf(ax1, sb[j * 4]), 0.f);
        float ih = fmaxf(fminf(ay2, sb[j * 4 + 3]) - fmaxf(ay1, sb[j * 4 + 1]), 0.f);
        float inter = iw * ih;
        float iou = inter / (areaA + areaB - inter + 1e-9f);
        if (iou > 0.7f) skeep[j] = 0;
      }
    }
  }
  __syncthreads();
  for (int j = tid; j < GTOT; j += blockDim.x) {
    if (skeep[j] && j < NOUT) {
      int g = sgid[j];
      float b0 = sb[j * 4], b1 = sb[j * 4 + 1], b2 = sb[j * 4 + 2], b3 = sb[j * 4 + 3];
      float r0 = resReg[g * 4], r1 = resReg[g * 4 + 1], r2 = resReg[g * 4 + 2], r3 = resReg[g * 4 + 3];
      float ww = b2 - b0, hh = b3 - b1;
      float x1 = b0 + r0 * ww, y1 = b1 + r1 * hh, x2 = b2 + r2 * ww, y2 = b3 + r3 * hh;
      float w2 = x2 - x1, h2 = y2 - y1;
      float mm = fmaxf(w2, h2);
      float q0 = x1 + w2 * 0.5f - mm * 0.5f;
      float q1 = y1 + h2 * 0.5f - mm * 0.5f;
      float q2 = q0 + mm, q3 = q1 + mm;
      int i0 = (int)q0, i1 = (int)q1, i2 = (int)q2, i3 = (int)q3;
      if ((i2 - i0 > 0) && (i3 - i1 > 0)) {
        out[j * 4 + 0] = q0; out[j * 4 + 1] = q1; out[j * 4 + 2] = q2; out[j * 4 + 3] = q3;
        out[(size_t)NOUT * 4 + j] = ssc[j];
        out[(size_t)NOUT * 5 + j] = 1.0f;
      }
    }
  }
}

// ---------------------------------------------------------------------------
extern "C" void kernel_launch(void* const* d_in, const int* in_sizes, int n_in,
                              void* d_out, int out_size, void* d_ws, size_t ws_size,
                              hipStream_t stream) {
  (void)in_sizes; (void)n_in; (void)ws_size;
  const float* img = (const float*)d_in[0];
  const float* w1 = (const float*)d_in[1];
  const float* b1 = (const float*)d_in[2];
  const float* a1 = (const float*)d_in[3];
  const float* w2 = (const float*)d_in[4];
  const float* b2 = (const float*)d_in[5];
  const float* a2 = (const float*)d_in[6];
  const float* w3 = (const float*)d_in[7];
  const float* b3 = (const float*)d_in[8];
  const float* a3 = (const float*)d_in[9];
  const float* wa = (const float*)d_in[10];
  const float* ba = (const float*)d_in[11];
  const float* wb = (const float*)d_in[12];
  const float* bbv = (const float*)d_in[13];
  float* out = (float*)d_out;

  // ---- scale pyramid (match Python double-precision semantics) ----
  const int H = 1080, W = 1920;
  double mfac = 12.0 / 20.0, minl = (double)H * mfac, f = 1.0;
  int nsc = 0;
  double sv[32]; int OH[32], OW[32], PH[32], PW[32], C2H[32], C2W[32], C3H[32], C3W[32];
  int Npix[32], Ksel[32], KKsel[32];
  int maxPool = 0, maxC2 = 0, maxN = 0;
  while (minl >= 12.0 && nsc < 32) {
    double s = mfac * f;
    int oh = (int)(H * s), ow = (int)(W * s);
    int c1h = oh - 2, c1w = ow - 2, ph = c1h / 2, pw = c1w / 2;
    int c2h = ph - 2, c2w = pw - 2, c3h = c2h - 2, c3w = c2w - 2;
    if (c3h > 0 && c3w > 0) {
      sv[nsc] = s; OH[nsc] = oh; OW[nsc] = ow; PH[nsc] = ph; PW[nsc] = pw;
      C2H[nsc] = c2h; C2W[nsc] = c2w; C3H[nsc] = c3h; C3W[nsc] = c3w;
      int n = c3h * c3w;
      Npix[nsc] = n;
      Ksel[nsc] = n < 512 ? n : 512;
      KKsel[nsc] = Ksel[nsc] < 128 ? Ksel[nsc] : 128;
      if (ph * pw > maxPool) maxPool = ph * pw;
      if (c2h * c2w > maxC2) maxC2 = c2h * c2w;
      if (n > maxN) maxN = n;
      nsc++;
    }
    minl *= 0.709; f *= 0.709;
  }
  int NOUT = 0;
  for (int i = 0; i < nsc; i++) NOUT += KKsel[i];
  if (NOUT * 6 > out_size) NOUT = out_size / 6;

  // ---- workspace layout ----
  size_t off = 0;
  char* wsb = (char*)d_ws;
  auto alloc = [&](size_t bytes) -> char* {
    char* p = wsb + off;
    off = (off + bytes + 255) & ~(size_t)255;
    return p;
  };
  _Float16* P    = (_Float16*)alloc((size_t)10 * maxPool * 2);
  _Float16* C2b  = (_Float16*)alloc((size_t)16 * maxC2 * 2);
  float*    prob = (float*)alloc((size_t)maxN * 4);
  float*    regb = (float*)alloc((size_t)maxN * 16);
  unsigned long long* keysA = (unsigned long long*)alloc(192 * 1024);
  unsigned long long* keysB = (unsigned long long*)alloc(192 * 1024);
  unsigned long long* topk  = (unsigned long long*)alloc(512 * 8);
  int*   resFlag  = (int*)alloc(GTOT * 4);
  float* resScore = (float*)alloc(GTOT * 4);
  float* resBox   = (float*)alloc((size_t)GTOT * 16);
  float* resReg   = (float*)alloc((size_t)GTOT * 16);
  unsigned long long* gkeys = (unsigned long long*)alloc(GTOT * 8);
  _Float16* b2sw = (_Float16*)alloc(3 * 512 * 2);
  _Float16* b3sw = (_Float16*)alloc(10 * 512 * 2);   // contiguous with bhsw:
  _Float16* bhsw = (_Float16*)alloc(512 * 2);        // 11264B TDM blob from b3sw

  // ---- init ----
  k_zero<<<(out_size + 255) / 256, 256, 0, stream>>>(out, out_size);
  k_zero<<<(GTOT + 255) / 256, 256, 0, stream>>>((float*)resFlag, GTOT);
  k_prep<<<28, 256, 0, stream>>>(w2, w3, wa, wb, b2sw, b3sw, bhsw);

  // ---- per-scale pipeline ----
  int scbase = 0;
  for (int sc = 0; sc < nsc; sc++) {
    int oh = OH[sc], ow = OW[sc], ph = PH[sc], pw = PW[sc];
    int c2h = C2H[sc], c2w = C2W[sc], c3h = C3H[sc], c3w = C3W[sc];
    int n = Npix[sc], K = Ksel[sc], KK = KKsel[sc];
    float rh = (float)((double)H / (double)oh);
    float rw = (float)((double)W / (double)ow);

    k_conv1pool<<<(ph * pw + 127) / 128, 128, 0, stream>>>(img, w1, b1, a1, P, ph, pw, rh, rw);

    int ntx2 = (c2w + 15) / 16;
    int tiles2 = ntx2 * c2h;
    k_conv2<<<(tiles2 + 7) / 8, 256, 0, stream>>>(P, b2sw, b2, a2, C2b, ph, pw, c2h, c2w, ntx2);

    int ntx3 = (c3w + 15) / 16;
    int tiles3 = ntx3 * c3h;
    k_conv3heads<<<(tiles3 + 7) / 8, 256, 0, stream>>>(C2b, b3sw, b3, a3, ba, bbv,
                                                       prob, regb, c2h, c2w, c3h, c3w, ntx3);

    // hierarchical exact top-K selection
    const float* srcP = prob;
    const unsigned long long* srcK = nullptr;
    int mcur = n, pb = 0;
    while (mcur > TKN) {
      int chunks = (mcur + TKN - 1) / TKN;
      unsigned long long* dst = pb ? keysB : keysA;
      k_topk_stage<<<chunks, 1024, 0, stream>>>(srcP, srcK, dst, mcur);
      srcP = nullptr; srcK = dst; pb ^= 1; mcur = chunks * 512;
    }
    k_topk_stage<<<1, 1024, 0, stream>>>(srcP, srcK, topk, mcur);

    k_nms_scale<<<1, 512, 0, stream>>>(topk, regb, (float)sv[sc], c3w, K, KK,
                                       scbase, resFlag, resScore, resBox, resReg);
    scbase += 128;  // fixed per-scale stride; slot order preserves global tiebreaks
  }

  // ---- global stage ----
  k_gsort<<<1, 1024, 0, stream>>>(resFlag, resScore, gkeys);
  k_gnms<<<1, 1024, 0, stream>>>(gkeys, resBox, resReg, out, NOUT);
}
